// RefGroupedQueryAttention_2714419331561
// MI455X (gfx1250) — matrix operand order repaired and checked by reading, not compiled, not run on hardware
//
#include <hip/hip_runtime.h>
#include <hip/hip_bf16.h>

#define SEQ  2048
#define EMB  2048
#define NH   16
#define NKV  8
#define HD   128
#define DOUT 2048

typedef __bf16 bf16_t;
typedef __attribute__((ext_vector_type(8)))  bf16_t bf16x8;
typedef __attribute__((ext_vector_type(16))) bf16_t bf16x16;
typedef __attribute__((ext_vector_type(8)))  float  f32x8;

static __device__ __forceinline__ bf16x16 join16(bf16x8 a, bf16x8 b) {
  return __builtin_shufflevector(a, b, 0,1,2,3,4,5,6,7,8,9,10,11,12,13,14,15);
}

static __device__ __forceinline__ f32x8 wmma_bf16(bf16x16 a, bf16x16 b, f32x8 c) {
  // (neg_a, A, neg_b, B, c_mod, C, reuse_a, reuse_b)
  return __builtin_amdgcn_wmma_f32_16x16x32_bf16(false, a, false, b, (short)0, c, false, false);
}

// CDNA5 async copy: global -> LDS without touching VGPR data path (ASYNCcnt).
// Generic LDS pointers carry the LDS byte offset in addr[31:0].
static __device__ __forceinline__ void async_ld_b128(void* lds, const void* gaddr) {
  unsigned l = (unsigned)(size_t)lds;
  asm volatile("global_load_async_to_lds_b128 %0, %1, off"
               :: "v"(l), "v"(gaddr) : "memory");
}
#define WAIT_ASYNC(n) asm volatile("s_wait_asynccnt %0" :: "n"(n) : "memory")

// ---------------------------------------------------------------------------
// elementwise f32 -> bf16
// ---------------------------------------------------------------------------
__global__ void f32_to_bf16_kern(const float* __restrict__ src,
                                 bf16_t* __restrict__ dst, int n) {
  int i = blockIdx.x * blockDim.x + threadIdx.x;
  if (i < n) dst[i] = (bf16_t)src[i];
}

// ---------------------------------------------------------------------------
// C[M,N] = A[M,K] @ B[N,K]^T + bias   (A,B bf16 row-major, C f32)
// 256 thr (8 waves in a 2x4 grid, 32x64 sub-tile each); block tile 128x128,
// K-step 32, double-buffered LDS staged with global_load_async_to_lds_b128
// so HBM fetch of tile i+1 overlaps WMMA compute on tile i.
// ---------------------------------------------------------------------------
__global__ __launch_bounds__(256) void gemm_bf16_kern(
    const bf16_t* __restrict__ A, const bf16_t* __restrict__ B,
    const float* __restrict__ bias, float* __restrict__ C,
    int M, int N, int K)
{
  __shared__ __align__(16) bf16_t As[2][128][40];
  __shared__ __align__(16) bf16_t Bs[2][128][40];

  const int tid  = threadIdx.x;
  const int lane = tid & 31;
  const int wv   = tid >> 5;
  const int wm   = wv & 1;            // wave row   (2 x 32 rows)
  const int wn   = wv >> 1;           // wave col   (4 x 64 cols)
  const int half = lane >> 4;
  const int lr   = lane & 15;
  const int bm   = blockIdx.y * 128;
  const int bn   = blockIdx.x * 128;

  f32x8 acc[2][4];
  f32x8 zero = {};
  for (int a = 0; a < 2; ++a)
    for (int b = 0; b < 4; ++b) acc[a][b] = zero;

  const int rowL = tid >> 2;          // 0..63
  const int colL = (tid & 3) * 8;     // 0,8,16,24

  // stage one 128x32 A tile + 128x32 B tile into buffer `buf` (4 async/thread)
  auto stage = [&](int buf, int kt) {
    for (int p = 0; p < 2; ++p) {
      int r = rowL + p * 64;
      async_ld_b128(&As[buf][r][colL], A + (size_t)(bm + r) * K + kt + colL);
      async_ld_b128(&Bs[buf][r][colL], B + (size_t)(bn + r) * K + kt + colL);
    }
  };

  const int kbA = half ? 16 : 0;      // A-frag k-offset
  const int kbB = half ? 16 : 0;      // B-frag k-offset

  auto compute = [&](int cur) {
    bf16x16 af[2];
    for (int a = 0; a < 2; ++a) {
      int am = wm * 32 + a * 16 + lr;
      af[a] = join16(*(const bf16x8*)(&As[cur][am][kbA]),
                     *(const bf16x8*)(&As[cur][am][kbA + 8]));
    }
    for (int b = 0; b < 4; ++b) {
      int bro = wn * 64 + b * 16 + lr;
      bf16x16 bf = join16(*(const bf16x8*)(&Bs[cur][bro][kbB]),
                          *(const bf16x8*)(&Bs[cur][bro][kbB + 8]));
      acc[0][b] = wmma_bf16(af[0], bf, acc[0][b]);
      acc[1][b] = wmma_bf16(af[1], bf, acc[1][b]);
    }
  };

  const int nk = K / 32;
  stage(0, 0);
  for (int it = 0; it < nk - 1; ++it) {          // branch-free steady state
    int cur = it & 1;
    stage(cur ^ 1, (it + 1) * 32);               // prefetch next tile
    WAIT_ASYNC(4);                               // wait only for current tile
    __syncthreads();
    compute(cur);
    __syncthreads();
  }
  WAIT_ASYNC(0);                                 // epilogue: last tile
  __syncthreads();
  compute((nk - 1) & 1);

  for (int b = 0; b < 4; ++b) {
    int col = bn + wn * 64 + b * 16 + lr;
    float bvv = bias ? bias[col] : 0.0f;
    for (int a = 0; a < 2; ++a) {
      int rowbase = bm + wm * 32 + a * 16 + half * 8;
      for (int r = 0; r < 8; ++r)
        C[(size_t)(rowbase + r) * N + col] = acc[a][b][r] + bvv;
    }
  }
}

// ---------------------------------------------------------------------------
// RMSNorm (over HD) + RoPE, f32 [T][H*128] -> bf16 [H][T][128]
// ---------------------------------------------------------------------------
__global__ __launch_bounds__(256) void rms_rope_kern(
    const float* __restrict__ src, const float* __restrict__ scale,
    const float* __restrict__ cosT, const float* __restrict__ sinT,
    const int* __restrict__ start_pos, bf16_t* __restrict__ dst,
    int H, int T)
{
  int wid  = (blockIdx.x * blockDim.x + threadIdx.x) >> 5;
  int lane = threadIdx.x & 31;
  if (wid >= T * H) return;
  int t = wid / H, h = wid % H;

  const float4 xv = *(const float4*)(src + (size_t)t * H * HD + h * HD + lane * 4);
  float ss = xv.x * xv.x + xv.y * xv.y + xv.z * xv.z + xv.w * xv.w;
  for (int off = 16; off >= 1; off >>= 1) ss += __shfl_xor(ss, off, 32);
  float inv = rsqrtf(ss * (1.0f / 128.0f) + 1e-6f);

  int trow = t + start_pos[0];
  const float4 cv = *(const float4*)(cosT + (size_t)trow * HD + lane * 4);
  const float4 sv = *(const float4*)(sinT + (size_t)trow * HD + lane * 4);
  const float4 gv = *(const float4*)(scale + lane * 4);

  float y[4] = { xv.x * inv * gv.x, xv.y * inv * gv.y,
                 xv.z * inv * gv.z, xv.w * inv * gv.w };
  float c[4] = { cv.x, cv.y, cv.z, cv.w };
  float s[4] = { sv.x, sv.y, sv.z, sv.w };

  bool low = (lane < 16);               // d < 64
  bf16_t* out = dst + ((size_t)h * T + t) * HD + lane * 4;
  for (int i = 0; i < 4; ++i) {
    float p = __shfl_xor(y[i], 16, 32); // partner at d ^ 64
    float rot = low ? -p : p;
    out[i] = (bf16_t)(y[i] * c[i] + rot * s[i]);
  }
}

// f32 [T][H*128] -> bf16 [H][T][128]  (for V)
__global__ void reshape_bf16_kern(const float* __restrict__ src,
                                  bf16_t* __restrict__ dst, int H, int T) {
  int i = blockIdx.x * blockDim.x + threadIdx.x;
  int n = T * H * HD;
  if (i >= n) return;
  int d = i & (HD - 1);
  int rest = i >> 7;
  int h = rest % H;
  int t = rest / H;
  dst[((size_t)h * T + t) * HD + d] = (bf16_t)src[(size_t)t * H * HD + h * HD + d];
}

// ---------------------------------------------------------------------------
// Flash-style causal attention.  Q bf16 [16][T][128], K/V bf16 [8][T][128]
// Ctx bf16 [T][2048].  grid = (T/128, NH), block 256 (8 waves x 16 q-rows).
// K chunks staged via async-to-LDS; V transposed through VGPRs while staging.
// ---------------------------------------------------------------------------
__global__ __launch_bounds__(256) void attn_kern(
    const bf16_t* __restrict__ Q, const bf16_t* __restrict__ Kh,
    const bf16_t* __restrict__ Vh, bf16_t* __restrict__ Ctx, int T)
{
  __shared__ __align__(16) bf16_t Ks[32][136];     // [kv][d]
  __shared__ __align__(16) bf16_t Vs[128][40];     // [d][kv] (transposed)
  __shared__ __align__(16) bf16_t Ps[8][16][40];   // per-wave P tile [q][kv]

  const int h  = blockIdx.y;
  const int qb = blockIdx.x;
  const int g  = h >> 1;                           // GROUP_SIZE = 2
  const int tid  = threadIdx.x;
  const int lane = tid & 31;
  const int wv   = tid >> 5;
  const int half = lane >> 4;
  const int lr   = lane & 15;
  const int q0   = qb * 128 + wv * 16;

  // preload Q A-fragments (4 x 16x32)
  bf16x16 qa[4];
  const bf16_t* qrow = Q + ((size_t)h * T + q0 + lr) * HD;
  for (int dc = 0; dc < 4; ++dc) {
    int kb = dc * 32 + (half ? 16 : 0);
    qa[dc] = join16(*(const bf16x8*)(qrow + kb),
                    *(const bf16x8*)(qrow + kb + 8));
  }

  f32x8 O[8];
  f32x8 zero = {};
  for (int j = 0; j < 8; ++j) O[j] = zero;
  float rm[8], rl[8];
  for (int r = 0; r < 8; ++r) { rm[r] = -1e30f; rl[r] = 0.0f; }

  const float sscale = 0.08838834764831845f;       // 1/sqrt(128)
  const int nch = (qb + 1) * 4;                    // 32-kv chunks up to diag

  for (int ch = 0; ch < nch; ++ch) {
    int kv0 = ch * 32;
    __syncthreads();
    {
      int r = tid >> 3;                 // 0..31
      int c = (tid & 7) * 16;           // 0..112
      const bf16_t* kp = Kh + ((size_t)g * T + kv0 + r) * HD + c;
      async_ld_b128(&Ks[r][c],     kp);
      async_ld_b128(&Ks[r][c + 8], kp + 8);
      const bf16_t* vp = Vh + ((size_t)g * T + kv0 + r) * HD + c;
      bf16x8 w0 = *(const bf16x8*)(vp);
      bf16x8 w1 = *(const bf16x8*)(vp + 8);
      for (int i = 0; i < 8; ++i) Vs[c + i][r]     = w0[i];
      for (int i = 0; i < 8; ++i) Vs[c + 8 + i][r] = w1[i];
      WAIT_ASYNC(0);
    }
    __syncthreads();

    // scores: two 16x16 tiles over d = 128, fragments batched ahead of WMMA
    f32x8 s0 = zero, s1 = zero;
    int kb2 = half ? 16 : 0;
    for (int dc = 0; dc < 4; dc += 2) {
      int dbA = dc * 32 + kb2;
      int dbB = (dc + 1) * 32 + kb2;
      bf16x16 k0a = join16(*(const bf16x8*)(&Ks[lr][dbA]),
                           *(const bf16x8*)(&Ks[lr][dbA + 8]));
      bf16x16 k1a = join16(*(const bf16x8*)(&Ks[16 + lr][dbA]),
                           *(const bf16x8*)(&Ks[16 + lr][dbA + 8]));
      bf16x16 k0b = join16(*(const bf16x8*)(&Ks[lr][dbB]),
                           *(const bf16x8*)(&Ks[lr][dbB + 8]));
      bf16x16 k1b = join16(*(const bf16x8*)(&Ks[16 + lr][dbB]),
                           *(const bf16x8*)(&Ks[16 + lr][dbB + 8]));
      s0 = wmma_bf16(qa[dc], k0a, s0);
      s1 = wmma_bf16(qa[dc], k1a, s1);
      s0 = wmma_bf16(qa[dc + 1], k0b, s0);
      s1 = wmma_bf16(qa[dc + 1], k1b, s1);
    }

    // scale + causal mask + online softmax
    float corr[8];
    for (int r = 0; r < 8; ++r) {
      int qg = q0 + half * 8 + r;
      float a0 = s0[r] * sscale;
      float a1 = s1[r] * sscale;
      if (kv0 + lr > qg)      a0 = -1e30f;
      if (kv0 + 16 + lr > qg) a1 = -1e30f;
      float tmx = fmaxf(a0, a1);
      tmx = fmaxf(tmx, __shfl_xor(tmx, 1, 32));
      tmx = fmaxf(tmx, __shfl_xor(tmx, 2, 32));
      tmx = fmaxf(tmx, __shfl_xor(tmx, 4, 32));
      tmx = fmaxf(tmx, __shfl_xor(tmx, 8, 32));
      float m2 = fmaxf(rm[r], tmx);
      corr[r] = __expf(rm[r] - m2);
      rm[r] = m2;
      float p0 = __expf(a0 - m2);
      float p1 = __expf(a1 - m2);
      s0[r] = p0; s1[r] = p1;
      float rs = p0 + p1;
      rs += __shfl_xor(rs, 1, 32);
      rs += __shfl_xor(rs, 2, 32);
      rs += __shfl_xor(rs, 4, 32);
      rs += __shfl_xor(rs, 8, 32);
      rl[r] = rl[r] * corr[r] + rs;
    }

    // stage P (C layout -> LDS -> A-fragment layout)
    for (int r = 0; r < 8; ++r) {
      int m = half * 8 + r;
      Ps[wv][m][lr]      = (bf16_t)s0[r];
      Ps[wv][m][16 + lr] = (bf16_t)s1[r];
    }
    int kbp = half ? 16 : 0;
    bf16x16 pfrag = join16(*(const bf16x8*)(&Ps[wv][lr][kbp]),
                           *(const bf16x8*)(&Ps[wv][lr][kbp + 8]));

    for (int j = 0; j < 8; ++j)
      for (int r = 0; r < 8; ++r) O[j][r] *= corr[r];

    int kbv = half ? 16 : 0;
    for (int j0 = 0; j0 < 8; j0 += 4) {
      bf16x16 vfr[4];
      for (int j = 0; j < 4; ++j) {
        int d = (j0 + j) * 16 + lr;
        vfr[j] = join16(*(const bf16x8*)(&Vs[d][kbv]),
                        *(const bf16x8*)(&Vs[d][kbv + 8]));
      }
      for (int j = 0; j < 4; ++j)
        O[j0 + j] = wmma_bf16(pfrag, vfr[j], O[j0 + j]);
    }
  }

  for (int j = 0; j < 8; ++j) {
    int d = j * 16 + lr;
    for (int r = 0; r < 8; ++r) {
      int trow = q0 + half * 8 + r;
      float o = O[j][r] / rl[r];
      Ctx[(size_t)trow * DOUT + h * HD + d] = (bf16_t)o;
    }
  }
}

// ---------------------------------------------------------------------------
extern "C" void kernel_launch(void* const* d_in, const int* in_sizes, int n_in,
                              void* d_out, int out_size, void* d_ws, size_t ws_size,
                              hipStream_t stream) {
  const float* x     = (const float*)d_in[0];
  /* d_in[1] = mask (bool): causality computed analytically, unused */
  const float* cosT  = (const float*)d_in[2];
  const float* sinT  = (const float*)d_in[3];
  const float* Wq    = (const float*)d_in[4];
  const float* bq    = (const float*)d_in[5];
  const float* Wk    = (const float*)d_in[6];
  const float* bk    = (const float*)d_in[7];
  const float* Wv    = (const float*)d_in[8];
  const float* bv    = (const float*)d_in[9];
  const float* Wo    = (const float*)d_in[10];
  const float* qsc   = (const float*)d_in[11];
  const float* ksc   = (const float*)d_in[12];
  const int*   spos  = (const int*)d_in[13];
  float* out = (float*)d_out;

  char* ws = (char*)d_ws;
  bf16_t* Xb  = (bf16_t*)(ws + 0);              //  8 MiB  (also reused as Ctx)
  bf16_t* Wqb = (bf16_t*)(ws + 8388608);        //  8 MiB
  bf16_t* Wkb = (bf16_t*)(ws + 16777216);       //  4 MiB
  bf16_t* Wvb = (bf16_t*)(ws + 20971520);       //  4 MiB
  bf16_t* Wob = (bf16_t*)(ws + 25165824);       //  8 MiB
  float*  Qf  = (float* )(ws + 33554432);       // 16 MiB
  float*  Kf  = (float* )(ws + 50331648);       //  8 MiB
  float*  Vf  = (float* )(ws + 58720256);       //  8 MiB
  bf16_t* Qb  = (bf16_t*)(ws + 67108864);       //  8 MiB
  bf16_t* Kb  = (bf16_t*)(ws + 75497472);       //  4 MiB
  bf16_t* Vb  = (bf16_t*)(ws + 79691776);       //  4 MiB
  bf16_t* Cb  = Xb;  // x is dead after the QKV GEMMs -> reuse for ctx

  auto cvt = [&](const float* s, bf16_t* d, int n) {
    f32_to_bf16_kern<<<(n + 255) / 256, 256, 0, stream>>>(s, d, n);
  };
  cvt(x,  Xb,  SEQ * EMB);
  cvt(Wq, Wqb, DOUT * EMB);
  cvt(Wk, Wkb, NKV * HD * EMB);
  cvt(Wv, Wvb, NKV * HD * EMB);
  cvt(Wo, Wob, EMB * DOUT);

  // QKV projections
  gemm_bf16_kern<<<dim3(DOUT / 128, SEQ / 128), 256, 0, stream>>>(
      Xb, Wqb, bq, Qf, SEQ, DOUT, EMB);
  gemm_bf16_kern<<<dim3((NKV * HD) / 128, SEQ / 128), 256, 0, stream>>>(
      Xb, Wkb, bk, Kf, SEQ, NKV * HD, EMB);
  gemm_bf16_kern<<<dim3((NKV * HD) / 128, SEQ / 128), 256, 0, stream>>>(
      Xb, Wvb, bv, Vf, SEQ, NKV * HD, EMB);

  // RMSNorm + RoPE (Q, K) and V relayout, all to head-major bf16
  rms_rope_kern<<<(SEQ * NH) / 8, 256, 0, stream>>>(Qf, qsc, cosT, sinT, spos, Qb, NH, SEQ);
  rms_rope_kern<<<(SEQ * NKV) / 8, 256, 0, stream>>>(Kf, ksc, cosT, sinT, spos, Kb, NKV, SEQ);
  reshape_bf16_kern<<<(SEQ * NKV * HD + 255) / 256, 256, 0, stream>>>(Vf, Vb, NKV, SEQ);

  // causal flash attention -> ctx bf16 [T][2048]
  attn_kern<<<dim3(SEQ / 128, NH), 256, 0, stream>>>(Qb, Kb, Vb, Cb, SEQ);

  // output projection -> f32 d_out
  gemm_bf16_kern<<<dim3(EMB / 128, SEQ / 128), 256, 0, stream>>>(
      Cb, Wob, nullptr, out, SEQ, EMB, DOUT);
}
